// Attention_10771777978404
// MI455X (gfx1250) — matrix operand order repaired and checked by reading, not compiled, not run on hardware
//
#include <hip/hip_runtime.h>

#define DEVI __device__ __forceinline__

typedef __attribute__((ext_vector_type(8)))  float        v8f;
typedef __attribute__((ext_vector_type(16))) __bf16       v16bf;
typedef __attribute__((ext_vector_type(4)))  unsigned int u32x4;
typedef __attribute__((ext_vector_type(2)))  unsigned int u32x2;
typedef unsigned short u16;
typedef unsigned int   u32;

constexpr int Bz = 2, S = 2048, DIM = 2048, NH = 16, HD = 128, RANK = 32;
constexpr int MROWS = Bz * S;          // 4096
constexpr float SCALE = 0.08838834764831845f; // 1/sqrt(128)

DEVI u16  f2bf(float x) { u32 u = __float_as_uint(x); return (u16)((u + 0x7FFFu + ((u >> 16) & 1u)) >> 16); }
DEVI float bf2f(u16 h)  { return __uint_as_float(((u32)h) << 16); }

union FragU { u32x4 q[2]; v16bf v; };

// A-matrix 16x32 bf16 fragment (ISA 7.12.2): lane<16 -> M=lane, K runs
// [koff..koff+7] and [koff+16..koff+23], koff = 8*(lane>=16).
DEVI v16bf load_a_frag(const u16* tile, int ld, int rbase, int kbase, int lane) {
  int row  = rbase + (lane & 15);
  int koff = kbase + ((lane >> 4) << 3);
  FragU f;
  f.q[0] = *reinterpret_cast<const u32x4*>(tile + row * ld + koff);
  f.q[1] = *reinterpret_cast<const u32x4*>(tile + row * ld + koff + 16);
  return f.v;
}
// B-matrix 32x16 bf16 fragment: lane holds column N=lane%16, K runs
// [koff..koff+7] and [koff+16..koff+23], koff = 8*(lane>=16).  tile layout is [N][K].
DEVI v16bf load_b_frag(const u16* tile, int ld, int nbase, int kbase, int lane) {
  int row  = nbase + (lane & 15);
  int koff = kbase + ((lane >> 4) << 3);
  FragU f;
  f.q[0] = *reinterpret_cast<const u32x4*>(tile + row * ld + koff);
  f.q[1] = *reinterpret_cast<const u32x4*>(tile + row * ld + koff + 16);
  return f.v;
}

DEVI v8f wmma_bf16(v16bf a, v16bf b, v8f c) {
  return __builtin_amdgcn_wmma_f32_16x16x32_bf16(false, a, false, b, (short)0, c, false, false);
}

// CDNA5 async global->LDS copy of 16 bytes (GVS addressing: sgpr base +
// 32-bit per-lane byte offset), tracked by ASYNCcnt.
DEVI void async_b128(u32 lds_addr, u32 goff, const void* sbase) {
  asm volatile("global_load_async_to_lds_b128 %0, %1, %2"
               :: "v"(lds_addr), "v"(goff), "s"(sbase) : "memory");
}
DEVI void wait_async_8() { asm volatile("s_wait_asynccnt 8" ::: "memory"); }
DEVI void wait_async_0() { asm volatile("s_wait_asynccnt 0" ::: "memory"); }
DEVI u32 lds_off(const void* p) { return (u32)(uintptr_t)p; }

// ---------------------------------------------------------------- fp32->bf16
__global__ void cvt_kernel(const float* __restrict__ src, u16* __restrict__ dst, int n) {
  for (int i = blockIdx.x * blockDim.x + threadIdx.x; i < n; i += gridDim.x * blockDim.x)
    dst[i] = f2bf(src[i]);
}

// ------------------------------------------------- C = A * B^T (both K-major)
// A: M x K bf16 row-major, B: N x K bf16 row-major. 128x128 block, 8 waves,
// each wave 32x64 (2x4 WMMA tiles). KTILE=64, double-buffered LDS filled by
// async-to-LDS prefetch (ASYNCcnt).
template <bool OUTF32>
__global__ __launch_bounds__(256)
void gemm_bf16_nt(const u16* __restrict__ A, const u16* __restrict__ Bm,
                  void* __restrict__ Cout, int Mdim, int Ndim, int Kdim) {
  constexpr int LDT = 72;                       // 64 + pad, 144B rows (16B aligned)
  __shared__ u16 As[2][128 * LDT];
  __shared__ u16 Bs[2][128 * LDT];
  const int tid  = threadIdx.x;
  const int lane = tid & 31, wid = tid >> 5;
  const int wm = wid & 3, wn = wid >> 2;        // 4x2 wave grid
  const int mbase = blockIdx.y * 128, nbase = blockIdx.x * 128;

  v8f acc[2][4]{};

  const int srow = tid >> 1;                    // staging: 2 threads/row
  const int scol = (tid & 1) * 32;              // each thread: 4 x 16B chunks

  auto issue_tile = [&](int buf, int k0) {
#pragma unroll
    for (int c = 0; c < 4; ++c) {
      int col = scol + c * 8;
      async_b128(lds_off(&As[buf][srow * LDT + col]),
                 (u32)(((size_t)(mbase + srow) * Kdim + k0 + col) * 2), A);
      async_b128(lds_off(&Bs[buf][srow * LDT + col]),
                 (u32)(((size_t)(nbase + srow) * Kdim + k0 + col) * 2), Bm);
    }
  };

  const int nit = Kdim / 64;
  issue_tile(0, 0);
  for (int it = 0; it < nit; ++it) {
    const int buf = it & 1;
    bool more = (it + 1 < nit);
    if (more) issue_tile(buf ^ 1, (it + 1) * 64);  // prefetch next buffer
    if (more) wait_async_8(); else wait_async_0(); // in-order: current buf done
    __syncthreads();

#pragma unroll
    for (int kb = 0; kb < 2; ++kb) {
      v16bf af[2], bf[4];
#pragma unroll
      for (int t = 0; t < 2; ++t) af[t] = load_a_frag(As[buf], LDT, wm * 32 + t * 16, kb * 32, lane);
#pragma unroll
      for (int u = 0; u < 4; ++u) bf[u] = load_b_frag(Bs[buf], LDT, wn * 64 + u * 16, kb * 32, lane);
#pragma unroll
      for (int t = 0; t < 2; ++t)
#pragma unroll
        for (int u = 0; u < 4; ++u) acc[t][u] = wmma_bf16(af[t], bf[u], acc[t][u]);
    }
    __syncthreads();                             // buffer free before reuse
  }

  const int half = lane >> 4, coln = lane & 15;
#pragma unroll
  for (int t = 0; t < 2; ++t)
#pragma unroll
    for (int u = 0; u < 4; ++u)
#pragma unroll
      for (int e = 0; e < 8; ++e) {
        int m = mbase + wm * 32 + t * 16 + e + half * 8;
        int n = nbase + wn * 64 + u * 16 + coln;
        float val = acc[t][u][e];
        if (OUTF32) ((float*)Cout)[(size_t)m * Ndim + n] = val;
        else        ((u16*)Cout)[(size_t)m * Ndim + n]   = f2bf(val);
      }
}

// ------------------------------------------------------------------- RoPE
__global__ void rope_kernel(u16* __restrict__ t, const float* __restrict__ cosb,
                            const float* __restrict__ sinb, int total) {
  for (int i = blockIdx.x * blockDim.x + threadIdx.x; i < total; i += gridDim.x * blockDim.x) {
    int p = i & 63;
    int h = (i >> 6) & (NH - 1);
    int s = (i >> 10) & (S - 1);
    int b = i >> 21;
    size_t base = ((size_t)(b * S + s) * DIM) + h * HD + 2 * p;
    float re = bf2f(t[base]), im = bf2f(t[base + 1]);
    float c = cosb[s * 64 + p], sn = sinb[s * 64 + p];
    t[base]     = f2bf(re * c - im * sn);
    t[base + 1] = f2bf(re * sn + im * c);
  }
}

// ------------------------------------------------- rank-32 gate projections
__global__ void lowrank_kernel(const u16* __restrict__ xb, const float* __restrict__ wa,
                               u16* __restrict__ out) {
  int idx = blockIdx.x * blockDim.x + threadIdx.x;
  if (idx >= MROWS * RANK) return;
  int m = idx >> 5, r = idx & 31;
  const u16* xr = xb + (size_t)m * DIM;
  const float* wr = wa + (size_t)r * DIM;
  float acc = 0.f;
#pragma unroll 4
  for (int k = 0; k < DIM; k += 4) {
    acc += bf2f(xr[k]) * wr[k] + bf2f(xr[k + 1]) * wr[k + 1]
         + bf2f(xr[k + 2]) * wr[k + 2] + bf2f(xr[k + 3]) * wr[k + 3];
  }
  out[idx] = f2bf(acc);
}

// -------------------------------------- flash attention + post-softmax gate
// grid = (S/64, B*NH), block = 128 (4 waves). Wave w owns query rows
// qbase + 16w .. +15, keeps full 16x64 score strip + 16x128 O in registers.
// K / Hk chunk staging uses async-to-LDS, overlapped with the V transpose.
__global__ __launch_bounds__(128)
void attn_kernel(const u16* __restrict__ q, const u16* __restrict__ k,
                 const u16* __restrict__ v, const u16* __restrict__ hq,
                 const u16* __restrict__ hk, u16* __restrict__ outb) {
  constexpr int LDQ = 136, LDV = 72, LDH = 40, LDP = 72;
  __shared__ u16 Qs[64 * LDQ];
  __shared__ u16 Ks[64 * LDQ];
  __shared__ u16 Vts[128 * LDV];      // V transposed: [d][key]
  __shared__ u16 Hqs[64 * LDH];
  __shared__ u16 Hks[64 * LDH];
  __shared__ u16 Ps[4][16 * LDP];     // per-wave gated-P staging

  const int tid = threadIdx.x, lane = tid & 31, wid = tid >> 5;
  const int half = lane >> 4, coln = lane & 15;
  const int b = blockIdx.y / NH, h = blockIdx.y % NH;
  const int qbase = blockIdx.x * 64;

  // ---- load Q tile (64 x 128) and Hq tile (64 x 32)
#pragma unroll
  for (int j = 0; j < 8; ++j) {
    int c = tid + j * 128;            // 0..1023
    int row = c >> 4, col = (c & 15) * 8;
    *reinterpret_cast<u32x4*>(&Qs[row * LDQ + col]) =
      *reinterpret_cast<const u32x4*>(q + ((size_t)(b * S + qbase + row) * DIM) + h * HD + col);
  }
  {
    int row = tid >> 1, col = (tid & 1) * 16;
    const u32x4* g = reinterpret_cast<const u32x4*>(hq + (size_t)(b * S + qbase + row) * RANK + col);
    u32x4* l = reinterpret_cast<u32x4*>(&Hqs[row * LDH + col]);
    l[0] = g[0]; l[1] = g[1];
  }
  __syncthreads();

  float rowmax[8], rowsum[8];
  v8f oacc[8]{};
#pragma unroll
  for (int e = 0; e < 8; ++e) { rowmax[e] = -3.0e38f; rowsum[e] = 0.f; }

  for (int kbase = 0; kbase <= qbase; kbase += 64) {
    // ---- async-stage K chunk (64 x 128) and Hk chunk (64 x 32)
#pragma unroll
    for (int j = 0; j < 8; ++j) {
      int c = tid + j * 128;
      int row = c >> 4, col = (c & 15) * 8;
      async_b128(lds_off(&Ks[row * LDQ + col]),
                 (u32)((((size_t)(b * S + kbase + row) * DIM) + h * HD + col) * 2), k);
    }
    {
      int row = tid >> 1, col = (tid & 1) * 16;
#pragma unroll
      for (int c = 0; c < 2; ++c)
        async_b128(lds_off(&Hks[row * LDH + col + c * 8]),
                   (u32)(((size_t)(b * S + kbase + row) * RANK + col + c * 8) * 2), hk);
    }
    // ---- load V chunk transposed (d-major), overlaps with async fills
#pragma unroll
    for (int j = 0; j < 16; ++j) {
      int c = tid + j * 128;          // 0..2047, 4 bf16 each
      int row = c >> 5, dcol = (c & 31) * 4;
      u32x2 d2 = *reinterpret_cast<const u32x2*>(v + ((size_t)(b * S + kbase + row) * DIM) + h * HD + dcol);
      const u16* e4 = reinterpret_cast<const u16*>(&d2);
#pragma unroll
      for (int jj = 0; jj < 4; ++jj) Vts[(dcol + jj) * LDV + row] = e4[jj];
    }
    wait_async_0();
    __syncthreads();

    // ---- S = Q K^T  (16x64 strip per wave, 16 WMMA)
    v8f sacc[4]{};
#pragma unroll
    for (int ks = 0; ks < 4; ++ks) {
      v16bf af = load_a_frag(Qs, LDQ, wid * 16, ks * 32, lane);
#pragma unroll
      for (int nt = 0; nt < 4; ++nt)
        sacc[nt] = wmma_bf16(af, load_b_frag(Ks, LDQ, nt * 16, ks * 32, lane), sacc[nt]);
    }
    // ---- gate logits G = Hq Hk^T  (rank 32, 4 WMMA)
    v8f gacc[4]{};
    {
      v16bf af = load_a_frag(Hqs, LDH, wid * 16, 0, lane);
#pragma unroll
      for (int nt = 0; nt < 4; ++nt)
        gacc[nt] = wmma_bf16(af, load_b_frag(Hks, LDH, nt * 16, 0, lane), gacc[nt]);
    }

    // ---- online softmax (rows owned per-wave; reductions across 16 lanes)
#pragma unroll
    for (int e = 0; e < 8; ++e) {
      int qrow = qbase + wid * 16 + e + half * 8;
      float tmax = -3.0e38f;
#pragma unroll
      for (int nt = 0; nt < 4; ++nt) {
        int key = kbase + nt * 16 + coln;
        float sv = sacc[nt][e] * SCALE + ((key > qrow) ? -1.0e9f : 0.0f);
        sacc[nt][e] = sv;
        tmax = fmaxf(tmax, sv);
      }
#pragma unroll
      for (int mdist = 1; mdist < 16; mdist <<= 1)
        tmax = fmaxf(tmax, __shfl_xor(tmax, mdist, 32));
      float mnew  = fmaxf(rowmax[e], tmax);
      float alpha = __expf(rowmax[e] - mnew);
      rowmax[e] = mnew;
      float psum = 0.f;
#pragma unroll
      for (int nt = 0; nt < 4; ++nt) {
        float p = __expf(sacc[nt][e] - mnew);
        float g = 1.0f / (1.0f + __expf(-gacc[nt][e]));
        psum += p;                                   // denominator: ungated
        Ps[wid][(e + half * 8) * LDP + nt * 16 + coln] = f2bf(p * g);
      }
#pragma unroll
      for (int mdist = 1; mdist < 16; mdist <<= 1)
        psum += __shfl_xor(psum, mdist, 32);
      rowsum[e] = rowsum[e] * alpha + psum;
#pragma unroll
      for (int dt = 0; dt < 8; ++dt) oacc[dt][e] *= alpha;
    }

    // ---- O += P_gated @ V   (16 WMMA)
#pragma unroll
    for (int ks = 0; ks < 2; ++ks) {
      v16bf af = load_a_frag(&Ps[wid][0], LDP, 0, ks * 32, lane);
#pragma unroll
      for (int dt = 0; dt < 8; ++dt)
        oacc[dt] = wmma_bf16(af, load_b_frag(Vts, LDV, dt * 16, ks * 32, lane), oacc[dt]);
    }
    __syncthreads();
  }

  // ---- normalize and store (b, qrow, h*HD + d) as bf16
#pragma unroll
  for (int e = 0; e < 8; ++e) {
    float inv = 1.0f / rowsum[e];
    int qrow = qbase + wid * 16 + e + half * 8;
#pragma unroll
    for (int dt = 0; dt < 8; ++dt) {
      int d = dt * 16 + coln;
      outb[((size_t)(b * S + qrow) * DIM) + h * HD + d] = f2bf(oacc[dt][e] * inv);
    }
  }
}

// --------------------------------------------------------------------------
extern "C" void kernel_launch(void* const* d_in, const int* in_sizes, int n_in,
                              void* d_out, int out_size, void* d_ws, size_t ws_size,
                              hipStream_t stream) {
  const float* x    = (const float*)d_in[0];
  const float* fcos = (const float*)d_in[2];
  const float* fsin = (const float*)d_in[3];
  const float* wq   = (const float*)d_in[4];
  const float* wk   = (const float*)d_in[5];
  const float* wv   = (const float*)d_in[6];
  const float* wo   = (const float*)d_in[7];
  const float* wa_q = (const float*)d_in[8];
  const float* wa_k = (const float*)d_in[9];
  float* out = (float*)d_out;

  // workspace carve-up (bytes)
  char* ws = (char*)d_ws;
  size_t off = 0;
  auto take = [&](size_t bytes) { char* p = ws + off; off += (bytes + 255) & ~size_t(255); return p; };
  u16* xb   = (u16*)take((size_t)MROWS * DIM * 2);
  u16* wqb  = (u16*)take((size_t)DIM * DIM * 2);
  u16* wkb  = (u16*)take((size_t)DIM * DIM * 2);
  u16* wvb  = (u16*)take((size_t)DIM * DIM * 2);
  u16* wob  = (u16*)take((size_t)DIM * DIM * 2);
  u16* qb   = (u16*)take((size_t)MROWS * DIM * 2);
  u16* kb   = (u16*)take((size_t)MROWS * DIM * 2);
  u16* vb   = (u16*)take((size_t)MROWS * DIM * 2);
  u16* hqb  = (u16*)take((size_t)MROWS * RANK * 2);
  u16* hkb  = (u16*)take((size_t)MROWS * RANK * 2);
  u16* attn = (u16*)take((size_t)MROWS * DIM * 2);
  (void)ws_size;

  // 1) fp32 -> bf16
  cvt_kernel<<<2048, 256, 0, stream>>>(x,  xb,  MROWS * DIM);
  cvt_kernel<<<2048, 256, 0, stream>>>(wq, wqb, DIM * DIM);
  cvt_kernel<<<2048, 256, 0, stream>>>(wk, wkb, DIM * DIM);
  cvt_kernel<<<2048, 256, 0, stream>>>(wv, wvb, DIM * DIM);
  cvt_kernel<<<2048, 256, 0, stream>>>(wo, wob, DIM * DIM);

  // 2) QKV projections (WMMA GEMM, async double-buffered, bf16 out)
  dim3 ggrid(DIM / 128, MROWS / 128);
  gemm_bf16_nt<false><<<ggrid, 256, 0, stream>>>(xb, wqb, qb, MROWS, DIM, DIM);
  gemm_bf16_nt<false><<<ggrid, 256, 0, stream>>>(xb, wkb, kb, MROWS, DIM, DIM);
  gemm_bf16_nt<false><<<ggrid, 256, 0, stream>>>(xb, wvb, vb, MROWS, DIM, DIM);

  // 3) RoPE on q, k
  int rope_total = MROWS * NH * (HD / 2);
  rope_kernel<<<4096, 256, 0, stream>>>(qb, fcos, fsin, rope_total);
  rope_kernel<<<4096, 256, 0, stream>>>(kb, fcos, fsin, rope_total);

  // 4) rank-32 gate projections
  lowrank_kernel<<<(MROWS * RANK) / 256, 256, 0, stream>>>(xb, wa_q, hqb);
  lowrank_kernel<<<(MROWS * RANK) / 256, 256, 0, stream>>>(xb, wa_k, hkb);

  // 5) flash attention with post-softmax gate
  attn_kernel<<<dim3(S / 64, Bz * NH), 128, 0, stream>>>(qb, kb, vb, hqb, hkb, attn);

  // 6) output projection (fp32 out)
  gemm_bf16_nt<true><<<ggrid, 256, 0, stream>>>(attn, wob, out, MROWS, DIM, DIM);
}
